// DeformableCONV_369367187487
// MI455X (gfx1250) — matrix operand order repaired and checked by reading, not compiled, not run on hardware
//
#include <hip/hip_runtime.h>
#include <hip/hip_bf16.h>

// ---------------------------------------------------------------------------
// Deformable conv v2 (K=3, DIL=1, PAD=1), B=8, C=17, H=128, W=192.
// Memory-bound (~390 MB streamed -> ~17us at 23.3 TB/s). GEMM (1 GFLOP) is
// pushed to the WMMA pipe: v_wmma_f32_16x16x32_f16 (f16 A/B, f32 acc).
// ---------------------------------------------------------------------------

typedef __attribute__((ext_vector_type(16))) _Float16 v16h;
typedef __attribute__((ext_vector_type(8)))  _Float16 v8h;
typedef __attribute__((ext_vector_type(8)))  float    v8f;

union V16U { v16h v; v8h h[2]; };

namespace {
constexpr int Kc   = 3;
constexpr int PAD  = 1;
constexpr int Cc   = 17;   // in = out channels
constexpr int KK   = Kc * Kc;          // 9
constexpr int CKK  = Cc * KK;          // 153 (reduction length)
constexpr int KPAD = 160;              // 5 chunks of 32
constexpr int B    = 8;
constexpr int H    = 128;
constexpr int W    = 192;
constexpr int HW   = H * W;            // 24576
constexpr int NPIX = 128;              // pixels per block
constexpr int SSTR = 168;              // LDS row stride in halves (mult of 8 -> 16B aligned frags)
}

__global__ __launch_bounds__(256)
void dcn_wmma_kernel(const float* __restrict__ x,
                     const float* __restrict__ offsets,
                     const float* __restrict__ mask,
                     const float* __restrict__ weight,
                     const float* __restrict__ bias,
                     float* __restrict__ out)
{
    __shared__ __align__(16) _Float16 sS[NPIX * SSTR]; // sampled tile  [pix][k]  ~42 KB
    __shared__ __align__(16) _Float16 sW[32 * SSTR];   // weight tile   [o]  [k]  ~10.5 KB

    const int tid     = threadIdx.x;
    const int tilesPerImg = HW / NPIX;                 // 192
    const int b       = blockIdx.x / tilesPerImg;
    const int pixBase = (blockIdx.x % tilesPerImg) * NPIX;

    // ---- stage weights: w[o][k] = weight[o, c, ki, kj], k = c*9 + ki*3 + kj ----
    for (int i = tid; i < 32 * KPAD; i += 256) {
        const int o = i / KPAD, k = i % KPAD;
        float v = 0.0f;
        if (o < Cc && k < CKK) {
            const int c = k / KK, j = k % KK;
            v = weight[((o * Cc + c) * Kc + (j / Kc)) * Kc + (j % Kc)];
        }
        sW[o * SSTR + k] = (_Float16)v;
    }

    // ---- phase 1: bilinear deformable sampling (f32) -> f16 LDS tile ----
    const int pix = tid & (NPIX - 1);
    const int rs  = tid >> 7;                          // 0 or 1: which half of rows
    const int hw  = pixBase + pix;
    const int h   = hw / W;
    const int w   = hw % W;

    if (rs == 0) {
        #pragma unroll
        for (int k = CKK; k < KPAD; ++k) sS[pix * SSTR + k] = (_Float16)0.0f;
    }

    const float* xb = x + (size_t)b * Cc * HW;
    for (int ckk = rs; ckk < CKK; ckk += 2) {
        const int c = ckk / KK, j = ckk % KK;
        const int ki = j / Kc, kj = j % Kc;
        // coalesced: 16+ consecutive lanes hit consecutive hw for the same channel
        const float offY = offsets[((size_t)b * 2 * CKK + 2 * ckk    ) * HW + hw];
        const float offX = offsets[((size_t)b * 2 * CKK + 2 * ckk + 1) * HW + hw];
        const float m    = mask   [((size_t)b * CKK + ckk)            * HW + hw];

        const float py  = (float)(h - PAD + ki) + offY;
        const float px  = (float)(w - PAD + kj) + offX;
        const float y0f = floorf(py), x0f = floorf(px);
        const float wy1 = py - y0f,   wx1 = px - x0f;
        const float wy0 = 1.0f - wy1, wx0 = 1.0f - wx1;
        const int   y0  = (int)y0f,   x0  = (int)x0f;

        const float* xc = xb + c * HW;
        auto corner = [&](int yy, int xx) -> float {
            if (yy < 0 || yy >= H || xx < 0 || xx >= W) return 0.0f;
            return xc[yy * W + xx];                    // L2-resident (x = 13.4 MB << 192 MB)
        };
        const float v = corner(y0,     x0    ) * (wy0 * wx0)
                      + corner(y0,     x0 + 1) * (wy0 * wx1)
                      + corner(y0 + 1, x0    ) * (wy1 * wx0)
                      + corner(y0 + 1, x0 + 1) * (wy1 * wx1);
        sS[pix * SSTR + ckk] = (_Float16)(v * m);
    }
    __syncthreads();

    // ---- phase 2: per-wave 16-pixel tile GEMM via v_wmma_f32_16x16x32_f16 ----
    // D[32 (O padded), 16 (pix)] = Wt[32,160] x S[160,16], 5 K-chunks x 2 M-tiles.
    const int wave   = tid >> 5;                       // 8 waves -> 8 pixel tiles
    const int lane   = tid & 31;
    const int nCol   = lane & 15;                      // N (pixel) / M (row) index
    const int hiHalf = lane >> 4;                      // half-wave selector
    const int pixT   = wave * 16;

    v8f acc0 = {};
    v8f acc1 = {};

    #pragma unroll
    for (int kc = 0; kc < KPAD / 32; ++kc) {
        const int k0 = kc * 32;

        // B fragment (32x16 f16): column nCol; halves 0..7 = K kb..kb+7,
        // halves 8..15 = K kb+16..kb+23, kb = k0 + hiHalf*8 (matches A layout).
        V16U bf;
        const _Float16* bp = &sS[(pixT + nCol) * SSTR + k0 + hiHalf * 8];
        bf.h[0] = *(const v8h*)(bp);
        bf.h[1] = *(const v8h*)(bp + 16);

        // A fragments (16x32 f16): lane row M = nCol; halves 0..7 = K kb..kb+7,
        // halves 8..15 = K kb+16..kb+23, kb = k0 + hiHalf*8 (ISA 7.12.2 A layout).
        V16U a0, a1;
        const _Float16* ap0 = &sW[ nCol       * SSTR + k0 + hiHalf * 8];
        a0.h[0] = *(const v8h*)(ap0);
        a0.h[1] = *(const v8h*)(ap0 + 16);
        const _Float16* ap1 = &sW[(16 + nCol) * SSTR + k0 + hiHalf * 8];
        a1.h[0] = *(const v8h*)(ap1);
        a1.h[1] = *(const v8h*)(ap1 + 16);

        acc0 = __builtin_amdgcn_wmma_f32_16x16x32_f16(false, a0.v, false, bf.v,
                                                      (short)0, acc0, false, false);
        acc1 = __builtin_amdgcn_wmma_f32_16x16x32_f16(false, a1.v, false, bf.v,
                                                      (short)0, acc1, false, false);
    }

    // ---- store: D layout -> lane 0-15: M=r, lane 16-31: M=r+8; N = lane&15 ----
    const int oBase = hiHalf * 8;
    const int hwOut = pixBase + pixT + nCol;
    float* outB = out + (size_t)b * Cc * HW;
    #pragma unroll
    for (int r = 0; r < 8; ++r) {
        const int o = oBase + r;                       // 0..15, all < 17
        outB[(size_t)o * HW + hwOut] = acc0[r] + bias[o];
    }
    if (oBase == 0) {                                  // M-tile 1: only o = 16 is real
        outB[(size_t)16 * HW + hwOut] = acc1[0] + bias[16];
    }
}

extern "C" void kernel_launch(void* const* d_in, const int* in_sizes, int n_in,
                              void* d_out, int out_size, void* d_ws, size_t ws_size,
                              hipStream_t stream) {
    (void)in_sizes; (void)n_in; (void)out_size; (void)d_ws; (void)ws_size;
    const float* x       = (const float*)d_in[0];
    const float* offsets = (const float*)d_in[1];
    const float* mask    = (const float*)d_in[2];
    const float* weight  = (const float*)d_in[3];
    const float* bias    = (const float*)d_in[4];
    float* out           = (float*)d_out;

    const int blocks = B * (HW / NPIX);                // 1536
    dcn_wmma_kernel<<<blocks, 256, 0, stream>>>(x, offsets, mask, weight, bias, out);
}
